// GAT_28441273434406
// MI455X (gfx1250) — matrix operand order repaired and checked by reading, not compiled, not run on hardware
//
#include <hip/hip_runtime.h>
#include <stdint.h>

#define NNODES 50000
#define NEDGES 800000
#define DIM 256
#define HDIM 128
#define NEG_SLOPE 0.2f

typedef __attribute__((ext_vector_type(16))) __bf16 v16bf;
typedef __attribute__((ext_vector_type(8)))  float v8f;

union BfPack { unsigned int w[8]; v16bf v; };

__device__ __forceinline__ unsigned short f2bf(float f) {
  unsigned int u = __float_as_uint(f);
  u += 0x7FFFu + ((u >> 16) & 1u);   // round-to-nearest-even
  return (unsigned short)(u >> 16);
}

// monotone-increasing unsigned encoding of float (for atomicMax on floats)
__device__ __forceinline__ unsigned int fenc(float f) {
  unsigned int u = __float_as_uint(f);
  return (u & 0x80000000u) ? ~u : (u | 0x80000000u);
}
__device__ __forceinline__ float fdec(unsigned int u) {
  return (u & 0x80000000u) ? __uint_as_float(u & 0x7FFFFFFFu) : __uint_as_float(~u);
}

__global__ void cvt_bf16_kernel(const float* __restrict__ in,
                                unsigned short* __restrict__ out, int n) {
  int i = blockIdx.x * blockDim.x + threadIdx.x;
  if (i < n) out[i] = f2bf(in[i]);
}

// Repack row-major [K][N] fp32 B into a per-lane-CONTIGUOUS WMMA bf16 B layout:
// for tile (kt,nt), lane n, VGPR v the value is
//   B[kt*32 + 16*(n/16) + 2v + {0,1}][nt*16 + n%16]
// stored at Bp[ ((tile*32 + lane)*8 + v) ]  -> each lane loads 8 consecutive
// dwords (= two b128 loads) per k-step in the GEMM.
__global__ void pack_b_kernel(const float* __restrict__ B,
                              unsigned int* __restrict__ Bp, int K, int N) {
  int blk = blockIdx.x;
  int ntiles = N >> 4;
  int kt = blk / ntiles, nt = blk % ntiles;
  int v = threadIdx.x >> 5, lane = threadIdx.x & 31;
  int k0 = kt * 32 + ((lane >> 4) << 4) + (v << 1);
  int n  = (nt << 4) + (lane & 15);
  unsigned int lo = f2bf(B[(size_t)k0 * N + n]);
  unsigned int hi = f2bf(B[(size_t)(k0 + 1) * N + n]);
  Bp[(((size_t)blk * 32 + lane) * 8) + v] = lo | (hi << 16);
}

// C[M x N] = A[M x 256](bf16 row-major) * Bp(packed bf16), optional +bias, ReLU.
// One wave per 16x16 tile; block = 8 waves covers 128 columns; K=256 unrolled
// -> 8 chained v_wmma_f32_16x16x32_bf16 per wave; all operands via b128 loads.
template<int N, bool BIAS, bool RELU>
__global__ __launch_bounds__(256)
void gemm_wmma_kernel(const unsigned short* __restrict__ A,
                      const unsigned int* __restrict__ Bp,
                      const float* __restrict__ bias,
                      float* __restrict__ C) {
  const int wave = threadIdx.x >> 5;
  const int lane = threadIdx.x & 31;
  const int half = lane >> 4;       // lanes 16..31 hold the "+16 K / +8 M" half
  const int rlo  = lane & 15;
  const int ntile = blockIdx.y * 8 + wave;
  const int mbase = blockIdx.x * 16;
  const int ntiles = N >> 4;
  // A layout: lane (half,rlo) holds row (mbase+rlo), K chunk [b..b+15] (b = 16*half)
  const unsigned short* arow = A + (size_t)(mbase + rlo) * DIM + half * 16;

  v8f c = {0.f, 0.f, 0.f, 0.f, 0.f, 0.f, 0.f, 0.f};
#pragma unroll
  for (int kt = 0; kt < DIM / 32; ++kt) {
    BfPack a, b;
    const uint4* pa = reinterpret_cast<const uint4*>(arow + kt * 32);
    uint4 qa0 = pa[0];   // K offsets 0..7   (VGPRs 0..3)
    uint4 qa1 = pa[1];   // K offsets 8..15  (VGPRs 4..7), +16 bytes
    a.w[0] = qa0.x; a.w[1] = qa0.y; a.w[2] = qa0.z; a.w[3] = qa0.w;
    a.w[4] = qa1.x; a.w[5] = qa1.y; a.w[6] = qa1.z; a.w[7] = qa1.w;
    const uint4* pb = reinterpret_cast<const uint4*>(
        Bp + (((size_t)(kt * ntiles + ntile) * 32 + lane) * 8));
    uint4 qb0 = pb[0];
    uint4 qb1 = pb[1];
    b.w[0] = qb0.x; b.w[1] = qb0.y; b.w[2] = qb0.z; b.w[3] = qb0.w;
    b.w[4] = qb1.x; b.w[5] = qb1.y; b.w[6] = qb1.z; b.w[7] = qb1.w;
    c = __builtin_amdgcn_wmma_f32_16x16x32_bf16(false, a.v, false, b.v,
                                                (short)0, c, false, false);
  }
  // C/D layout: VGPR r -> row r (lanes 0-15) / row r+8 (lanes 16-31), col = lane%16
#pragma unroll
  for (int r = 0; r < 8; ++r) {
    int row = mbase + r + half * 8;
    int col = (ntile << 4) + rlo;
    float val = c[r];
    if (BIAS) val += bias[col];
    if (RELU) val = fmaxf(val, 0.f);
    C[(size_t)row * N + col] = val;
  }
}

// a_src[n] = h[n,:].att_src ; a_dst[n] = h[n,:].att_dst ; one wave32 per node
__global__ __launch_bounds__(256)
void attn_dots_kernel(const float* __restrict__ h, const float* __restrict__ w_src,
                      const float* __restrict__ w_dst, float* __restrict__ a_src,
                      float* __restrict__ a_dst) {
  int wave = threadIdx.x >> 5, lane = threadIdx.x & 31;
  int node = blockIdx.x * 8 + wave;
  if (node >= NNODES) return;
  const float* hr = h + (size_t)node * DIM;
  float ss = 0.f, sd = 0.f;
#pragma unroll
  for (int i = lane; i < DIM; i += 32) {
    float v = hr[i];
    ss += v * w_src[i];
    sd += v * w_dst[i];
  }
#pragma unroll
  for (int off = 16; off > 0; off >>= 1) {
    ss += __shfl_xor(ss, off, 32);
    sd += __shfl_xor(sd, off, 32);
  }
  if (lane == 0) { a_src[node] = ss; a_dst[node] = sd; }
}

__global__ void edge_logit_kernel(const int* __restrict__ ei,
                                  const float* __restrict__ a_src,
                                  const float* __restrict__ a_dst,
                                  float* __restrict__ e,
                                  unsigned int* __restrict__ menc) {
  int i = blockIdx.x * blockDim.x + threadIdx.x;
  if (i >= NEDGES) return;
  int s = ei[i], d = ei[NEDGES + i];
  float v = a_src[s] + a_dst[d];
  v = (v > 0.f) ? v : v * NEG_SLOPE;
  e[i] = v;
  atomicMax(&menc[d], fenc(v));
}

__global__ void edge_exp_kernel(const int* __restrict__ ei, float* __restrict__ e,
                                const unsigned int* __restrict__ menc,
                                float* __restrict__ denom) {
  int i = blockIdx.x * blockDim.x + threadIdx.x;
  if (i >= NEDGES) return;
  int d = ei[NEDGES + i];
  float p = __expf(e[i] - fdec(menc[d]));
  e[i] = p;
  atomicAdd(&denom[d], p);
}

// 64 threads per edge, float4 gather of h[src] (b128 loads) + f32 atomic scatter.
// block = 256 threads handles 4 edges.
__global__ __launch_bounds__(256)
void edge_agg_kernel(const int* __restrict__ ei, const float* __restrict__ p,
                     const float* __restrict__ denom, const float* __restrict__ h,
                     float* __restrict__ agg) {
  int eidx = blockIdx.x * 4 + (threadIdx.x >> 6);
  int t = threadIdx.x & 63;                       // element group: 4 floats each
  int s = ei[eidx], d = ei[NEDGES + eidx];
  float alpha = p[eidx] / (denom[d] + 1e-16f);
  float4 hv = reinterpret_cast<const float4*>(h + (size_t)s * DIM)[t];
  float* ap = agg + (size_t)d * DIM + t * 4;
  atomicAdd(ap + 0, alpha * hv.x);
  atomicAdd(ap + 1, alpha * hv.y);
  atomicAdd(ap + 2, alpha * hv.z);
  atomicAdd(ap + 3, alpha * hv.w);
}

__global__ void bias_relu_cvt_kernel(const float* __restrict__ agg,
                                     const float* __restrict__ bias,
                                     unsigned short* __restrict__ out) {
  int i = blockIdx.x * blockDim.x + threadIdx.x;   // NNODES*DIM threads
  float v = agg[i] + bias[i & (DIM - 1)];
  out[i] = f2bf(fmaxf(v, 0.f));
}

extern "C" void kernel_launch(void* const* d_in, const int* in_sizes, int n_in,
                              void* d_out, int out_size, void* d_ws, size_t ws_size,
                              hipStream_t stream) {
  const float* x       = (const float*)d_in[0];
  const int*   ei      = (const int*)d_in[1];    // [2][NEDGES]
  const float* W       = (const float*)d_in[2];
  const float* att_src = (const float*)d_in[3];
  const float* att_dst = (const float*)d_in[4];
  const float* bias    = (const float*)d_in[5];
  const float* Wp      = (const float*)d_in[6];
  const float* bp      = (const float*)d_in[7];
  float* out = (float*)d_out;

  char* ws = (char*)d_ws;
  size_t off = 0;
  auto alloc = [&](size_t bytes) -> void* {
    void* p = ws + off;
    off += (bytes + 255) & ~(size_t)255;
    return p;
  };
  unsigned short* x_bf  = (unsigned short*)alloc((size_t)NNODES * DIM * 2);
  unsigned int*   Wpk   = (unsigned int*)alloc((size_t)(DIM / 32) * (DIM / 16) * 256 * 4);
  unsigned int*   Wppk  = (unsigned int*)alloc((size_t)(DIM / 32) * (HDIM / 16) * 256 * 4);
  float*          h     = (float*)alloc((size_t)NNODES * DIM * 4);
  float*          a_src = (float*)alloc((size_t)NNODES * 4);
  float*          a_dst = (float*)alloc((size_t)NNODES * 4);
  float*          ep    = (float*)alloc((size_t)NEDGES * 4);
  unsigned int*   menc  = (unsigned int*)alloc((size_t)NNODES * 4);
  float*          denom = (float*)alloc((size_t)NNODES * 4);
  float*          agg   = (float*)alloc((size_t)NNODES * DIM * 4);
  unsigned short* t_bf  = (unsigned short*)alloc((size_t)NNODES * DIM * 2);

  hipMemsetAsync(menc, 0, (size_t)NNODES * 4, stream);
  hipMemsetAsync(denom, 0, (size_t)NNODES * 4, stream);
  hipMemsetAsync(agg, 0, (size_t)NNODES * DIM * 4, stream);

  // 1. convert x to bf16; repack W / Wp into WMMA B layout
  cvt_bf16_kernel<<<(NNODES * DIM) / 256, 256, 0, stream>>>(x, x_bf, NNODES * DIM);
  pack_b_kernel<<<(DIM / 32) * (DIM / 16), 256, 0, stream>>>(W, Wpk, DIM, DIM);
  pack_b_kernel<<<(DIM / 32) * (HDIM / 16), 256, 0, stream>>>(Wp, Wppk, DIM, HDIM);

  // 2. h = x @ W   (WMMA bf16 -> f32)
  gemm_wmma_kernel<DIM, false, false>
      <<<dim3(NNODES / 16, DIM / 128), 256, 0, stream>>>(x_bf, Wpk, nullptr, h);

  // 3. attention dots
  attn_dots_kernel<<<NNODES / 8, 256, 0, stream>>>(h, att_src, att_dst, a_src, a_dst);

  // 4. edge logits + segment max (encoded atomicMax)
  edge_logit_kernel<<<(NEDGES + 255) / 256, 256, 0, stream>>>(ei, a_src, a_dst, ep, menc);

  // 5. exp + segment sum
  edge_exp_kernel<<<(NEDGES + 255) / 256, 256, 0, stream>>>(ei, ep, menc, denom);

  // 6. scatter-aggregate alpha * h[src] into agg[dst]  (4 edges / block)
  edge_agg_kernel<<<NEDGES / 4, 256, 0, stream>>>(ei, ep, denom, h, agg);

  // 7. relu(agg + bias) -> bf16
  bias_relu_cvt_kernel<<<(NNODES * DIM) / 256, 256, 0, stream>>>(agg, bias, t_bf);

  // 8. out = relu(t @ Wp + bp)   (WMMA bf16 -> f32, fused epilogue)
  gemm_wmma_kernel<HDIM, true, true>
      <<<dim3(NNODES / 16, HDIM / 128), 256, 0, stream>>>(t_bf, Wppk, bp, out);
}
